// APPNP_35708358099577
// MI455X (gfx1250) — matrix operand order repaired and checked by reading, not compiled, not run on hardware
//
#include <hip/hip_runtime.h>

typedef __attribute__((ext_vector_type(16))) _Float16 v16h;
typedef __attribute__((ext_vector_type(8)))  float    v8f;
typedef int v4i __attribute__((vector_size(16)));

#define IN_F  512
#define HID   128
#define OUT_F 32

union Frag { v16h h; unsigned int u[8]; };

// hot-path pack: 1 VALU (v_cvt_pkrtz_f16_f32)
__device__ __forceinline__ unsigned int packh_rtz(float lo, float hi) {
  __fp16 p2[2];
  *(decltype(__builtin_amdgcn_cvt_pkrtz(0.f, 0.f))*)p2 = __builtin_amdgcn_cvt_pkrtz(lo, hi);
  unsigned int u; __builtin_memcpy(&u, p2, 4);
  return u;
}
// one-shot path: RNE via v_cvt_f16_f32
__device__ __forceinline__ unsigned int packh_rn(float lo, float hi) {
  _Float16 l = (_Float16)lo, h = (_Float16)hi;
  unsigned short ul, uh;
  __builtin_memcpy(&ul, &l, 2); __builtin_memcpy(&uh, &h, 2);
  return (unsigned int)ul | ((unsigned int)uh << 16);
}
// A-matrix (16x32 f16) K position for VGPR v, lane half (ISA 7.12.2)
__device__ __forceinline__ int akmap(int v, int laneHi) {
  return (v < 4 ? 2 * v : 16 + 2 * (v - 4)) + (laneHi ? 8 : 0);
}

// -------- async global->LDS staging (CDNA5), with sync fallback --------
__device__ __forceinline__ void cp_async_b128(const unsigned int* g, unsigned int* l) {
#if __has_builtin(__builtin_amdgcn_global_load_async_to_lds_b128)
  __builtin_amdgcn_global_load_async_to_lds_b128((v4i*)g, (v4i*)l, 0, 0);
#else
  *(uint4*)l = *(const uint4*)g;
#endif
}
__device__ __forceinline__ void wait_async_all() {
#if __has_builtin(__builtin_amdgcn_global_load_async_to_lds_b128)
#if __has_builtin(__builtin_amdgcn_s_wait_asynccnt)
  __builtin_amdgcn_s_wait_asynccnt(0);
#else
  asm volatile("s_wait_asynccnt 0" ::: "memory");
#endif
#endif
}

// One-shot: convert W1/W2 to f16 pairs in WMMA fragment-linear order.
// w1p flat idx = (((kh*8+kt)*8+nt)*32+lane)*8+v  (32768 dwords, 128KB)
// w2p flat idx = (((kt*2+nt))*32+lane)*8+v       (2048 dwords, 8KB)
__global__ __launch_bounds__(256)
void prep_weights(const float* __restrict__ W1, const float* __restrict__ W2,
                  unsigned int* __restrict__ w1p, unsigned int* __restrict__ w2p) {
  int t = blockIdx.x * blockDim.x + threadIdx.x;
  if (t < 32768) {
    int v = t & 7, lane = (t >> 3) & 31, nt = (t >> 8) & 7, kt = (t >> 11) & 7, kh = (t >> 14) & 1;
    int laneHi = lane >> 4, lm = lane & 15;
    int k = kh * 256 + kt * 32 + akmap(v, laneHi);
    int n = nt * 16 + lm;
    w1p[t] = packh_rn(W1[k * HID + n], W1[(k + 1) * HID + n]);
  }
  if (t < 2048) {
    int v = t & 7, lane = (t >> 3) & 31, nt = (t >> 8) & 1, kt = (t >> 9) & 3;
    int laneHi = lane >> 4, lm = lane & 15;
    int k = kt * 32 + akmap(v, laneHi);
    int n = nt * 16 + lm;
    w2p[t] = packh_rn(W2[k * OUT_F + n], W2[(k + 1) * OUT_F + n]);
  }
}

// Fused MLP: h0 = relu(relu(x@W1+b1)@W2+b2); f16 WMMA, weights async-staged
// to LDS, B-fragments are contiguous ds_load_b128 pairs.
__global__ __launch_bounds__(256)
void mlp_fused(const float* __restrict__ x,
               const unsigned int* __restrict__ w1p, const float* __restrict__ b1,
               const unsigned int* __restrict__ w2p, const float* __restrict__ b2,
               float* __restrict__ h0, int n_nodes)
{
  __shared__ unsigned int w1s[16384];       // 64KB: one K-half, fragment-linear
  __shared__ unsigned int w2s[2048];        // 8KB
  __shared__ _Float16     h1s[128 * HID];   // 32KB: relu(h1) tile, f16
  __shared__ float b1s[HID], b2s[OUT_F];

  const int tid    = threadIdx.x;
  const int wv     = tid >> 5;              // wave32
  const int lane   = tid & 31;
  const int laneHi = lane >> 4;
  const int lm     = lane & 15;
  const int m0     = blockIdx.x * 128 + wv * 16;

  if (tid < HID)   b1s[tid] = b1[tid];
  if (tid < OUT_F) b2s[tid] = b2[tid];
  #pragma unroll
  for (int q = tid; q < 512; q += 256)      // W2: 2048 dwords
    cp_async_b128(w2p + q * 4, w2s + q * 4);

  int rowA = m0 + lm;
  if (rowA > n_nodes - 1) rowA = n_nodes - 1;   // clamp loads; stores guarded
  const float* xr = x + (size_t)rowA * IN_F;

  v8f acc[8] = {};
  for (int kh = 0; kh < 2; ++kh) {
    __syncthreads();                        // prior reads of w1s done
    #pragma unroll
    for (int q = tid; q < 4096; q += 256)   // 64KB half of W1
      cp_async_b128(w1p + kh * 16384 + q * 4, w1s + q * 4);
    wait_async_all();
    __syncthreads();

    #pragma unroll
    for (int kt = 0; kt < 8; ++kt) {
      Frag a;
      #pragma unroll
      for (int v = 0; v < 8; v += 2) {
        int k = kh * 256 + kt * 32 + akmap(v, laneHi);
        float4 xv = *(const float4*)(xr + k);         // akmap: v,v+1 contiguous
        a.u[v]     = packh_rtz(xv.x, xv.y);
        a.u[v + 1] = packh_rtz(xv.z, xv.w);
      }
      const unsigned int* bp = w1s + (kt * 8) * 32 * 8 + lane * 8;
      #pragma unroll
      for (int ng = 0; ng < 2; ++ng) {      // groups of 4: overlap ds with wmma
        Frag b0, b1f, b2f, b3;
        __builtin_memcpy(b0.u,  bp + (ng * 4 + 0) * 256, 32);
        __builtin_memcpy(b1f.u, bp + (ng * 4 + 1) * 256, 32);
        __builtin_memcpy(b2f.u, bp + (ng * 4 + 2) * 256, 32);
        __builtin_memcpy(b3.u,  bp + (ng * 4 + 3) * 256, 32);
        acc[ng * 4 + 0] = __builtin_amdgcn_wmma_f32_16x16x32_f16(
            false, a.h, false, b0.h,  (short)0, acc[ng * 4 + 0], false, false);
        acc[ng * 4 + 1] = __builtin_amdgcn_wmma_f32_16x16x32_f16(
            false, a.h, false, b1f.h, (short)0, acc[ng * 4 + 1], false, false);
        acc[ng * 4 + 2] = __builtin_amdgcn_wmma_f32_16x16x32_f16(
            false, a.h, false, b2f.h, (short)0, acc[ng * 4 + 2], false, false);
        acc[ng * 4 + 3] = __builtin_amdgcn_wmma_f32_16x16x32_f16(
            false, a.h, false, b3.h,  (short)0, acc[ng * 4 + 3], false, false);
      }
    }
  }
  __syncthreads();
  // epilogue 1: bias + relu -> f16 tile in LDS (row-major; A of GEMM2)
  #pragma unroll
  for (int nt = 0; nt < 8; ++nt) {
    #pragma unroll
    for (int r = 0; r < 8; ++r) {
      float vv = acc[nt][r] + b1s[nt * 16 + lm];
      vv = vv > 0.f ? vv : 0.f;
      h1s[(wv * 16 + laneHi * 8 + r) * HID + nt * 16 + lm] = (_Float16)vv;
    }
  }
  __syncthreads();

  // GEMM2: [16 x 128] @ [128 x 32]
  v8f acc2[2] = {};
  #pragma unroll
  for (int kt = 0; kt < 4; ++kt) {
    Frag a;
    #pragma unroll
    for (int v = 0; v < 8; ++v) {
      int k = kt * 32 + akmap(v, laneHi);   // even -> dword-aligned, merges to b128
      a.u[v] = *(const unsigned int*)&h1s[(wv * 16 + lm) * HID + k];
    }
    Frag b0, b1f;
    __builtin_memcpy(b0.u,  w2s + ((kt * 2 + 0) * 32 + lane) * 8, 32);
    __builtin_memcpy(b1f.u, w2s + ((kt * 2 + 1) * 32 + lane) * 8, 32);
    acc2[0] = __builtin_amdgcn_wmma_f32_16x16x32_f16(
        false, a.h, false, b0.h,  (short)0, acc2[0], false, false);
    acc2[1] = __builtin_amdgcn_wmma_f32_16x16x32_f16(
        false, a.h, false, b1f.h, (short)0, acc2[1], false, false);
  }
  #pragma unroll
  for (int nt = 0; nt < 2; ++nt) {
    #pragma unroll
    for (int r = 0; r < 8; ++r) {
      int row = m0 + laneHi * 8 + r;
      if (row < n_nodes) {
        float vv = acc2[nt][r] + b2s[nt * 16 + lm];
        vv = vv > 0.f ? vv : 0.f;
        h0[(size_t)row * OUT_F + nt * 16 + lm] = vv;
      }
    }
  }
}

// h_next = alpha * h
__global__ void prop_init(const float4* __restrict__ h, float4* __restrict__ hn, int n4) {
  int i = blockIdx.x * blockDim.x + threadIdx.x;
  if (i < n4) {
    float4 v = h[i];
    hn[i] = make_float4(0.01f * v.x, 0.01f * v.y, 0.01f * v.z, 0.01f * v.w);
  }
}

// h_next[row] += 0.99*val*h[col]; 8 lanes/edge, float4/lane. State + edges live in L2.
__global__ void prop_edges(const long long* __restrict__ er, const long long* __restrict__ ec,
                           const float* __restrict__ ev, const float* __restrict__ h,
                           float* __restrict__ hn, int n_edges) {
  int t = blockIdx.x * blockDim.x + threadIdx.x;
  int e = t >> 3;
  if (e >= n_edges) return;
  int c = (t & 7) << 2;
  long long r  = er[e];
  long long cl = ec[e];
  float v = 0.99f * ev[e];
  const float4 hv = *(const float4*)(h + cl * OUT_F + c);
  float* o = hn + r * OUT_F + c;
  unsafeAtomicAdd(o + 0, v * hv.x);
  unsafeAtomicAdd(o + 1, v * hv.y);
  unsafeAtomicAdd(o + 2, v * hv.z);
  unsafeAtomicAdd(o + 3, v * hv.w);
}

extern "C" void kernel_launch(void* const* d_in, const int* in_sizes, int n_in,
                              void* d_out, int out_size, void* d_ws, size_t ws_size,
                              hipStream_t stream) {
  const float*     x  = (const float*)d_in[0];
  const long long* er = (const long long*)d_in[1];
  const long long* ec = (const long long*)d_in[2];
  const float*     ev = (const float*)d_in[3];
  const float*     W1 = (const float*)d_in[4];
  const float*     b1 = (const float*)d_in[5];
  const float*     W2 = (const float*)d_in[6];
  const float*     b2 = (const float*)d_in[7];
  float* out = (float*)d_out;

  const int n_nodes = in_sizes[0] / IN_F;
  const int n_edges = in_sizes[1];

  float* bufA = (float*)d_ws;
  float* bufB = bufA + (size_t)n_nodes * OUT_F;
  unsigned int* w1p = (unsigned int*)(bufB + (size_t)n_nodes * OUT_F);
  unsigned int* w2p = w1p + 32768;

  prep_weights<<<128, 256, 0, stream>>>(W1, W2, w1p, w2p);

  int mblocks = (n_nodes + 127) / 128;
  mlp_fused<<<mblocks, 256, 0, stream>>>(x, w1p, b1, w2p, b2, bufA, n_nodes);

  int n4 = (n_nodes * OUT_F) / 4;
  int iblocks = (n4 + 255) / 256;
  int eblocks = (int)(((long long)n_edges * 8 + 255) / 256);
  for (int k = 0; k < 10; ++k) {
    const float* cur = (k & 1) ? bufB : bufA;
    float* nxt = (k == 9) ? out : ((k & 1) ? bufA : bufB);
    prop_init <<<iblocks, 256, 0, stream>>>((const float4*)cur, (float4*)nxt, n4);
    prop_edges<<<eblocks, 256, 0, stream>>>(er, ec, ev, cur, nxt, n_edges);
  }
}
